// DMPNN_31963146616864
// MI455X (gfx1250) — matrix operand order repaired and checked by reading, not compiled, not run on hardware
//
#include <hip/hip_runtime.h>
#include <hip/hip_bf16.h>

// ---------------------------------------------------------------------------
// DMPNN-style message passing for MI455X (gfx1250, wave32).
// Dense node GEMMs (x@W_u, x@W_v) run on v_wmma_f32_16x16x32_f16; the rest is
// memory-bound streaming + f32 atomics, sized for 23.3 TB/s HBM.
// ---------------------------------------------------------------------------

#define DIM    128
#define EDIM   16
#define T_ITER 3
#define NGRAPH 4096   // num_graphs (reference constant; device copy in d_in[15])

typedef __attribute__((ext_vector_type(16))) _Float16 v16h;
typedef __attribute__((ext_vector_type(8)))  _Float16 v8h;
typedef __attribute__((ext_vector_type(8)))  float    v8f;

// Order-preserving float <-> uint mapping (for atomicMax on floats).
__device__ __forceinline__ unsigned fkey(float f) {
    unsigned u = __float_as_uint(f);
    return (u & 0x80000000u) ? ~u : (u | 0x80000000u);
}
__device__ __forceinline__ float fval(unsigned k) {
    unsigned u = (k & 0x80000000u) ? (k & 0x7fffffffu) : ~k;
    return __uint_as_float(u);
}
// fkey(-inf): bits(-inf)=0xFF800000, sign set -> ~u = 0x007FFFFF
#define FKEY_NEG_INF 0x007FFFFFu

// ---------------------------------------------------------------------------
// Utility kernels
// ---------------------------------------------------------------------------
__global__ void zero_f32(float* __restrict__ p, long n) {
    long i = (long)blockIdx.x * blockDim.x + threadIdx.x;
    if (i < n) p[i] = 0.0f;
}

__global__ void set_u32(unsigned* __restrict__ p, unsigned v, long n) {
    long i = (long)blockIdx.x * blockDim.x + threadIdx.x;
    if (i < n) p[i] = v;
}

__global__ void f32_to_f16(const float* __restrict__ in, _Float16* __restrict__ out, long n) {
    long i = (long)blockIdx.x * blockDim.x + threadIdx.x;
    if (i < n) out[i] = (_Float16)in[i];
}

// ---------------------------------------------------------------------------
// Node GEMM: Out[M,128] = A16[M,128] @ W16[128,128], f16 in / f32 accumulate.
// 256 threads = 8 waves; each wave owns a 16-row M-tile and all 128 output
// columns (8 N-tiles), accumulating over K=128 in 4 steps of 32.
// W is staged *transposed* in LDS so B fragments are contiguous 16B ds loads.
// ---------------------------------------------------------------------------
__global__ __launch_bounds__(256)
void node_gemm_wmma(const _Float16* __restrict__ A16,
                    const _Float16* __restrict__ W16,   // [K=128][N=128] row-major
                    float* __restrict__ Out,            // [M][128]
                    int M)
{
    __shared__ _Float16 Wt[DIM * DIM];  // transposed: Wt[n*128 + k]  (32 KB)

    const int tid = threadIdx.x;
    for (int i = tid; i < DIM * DIM; i += 256) {
        int k = i >> 7, n = i & 127;
        Wt[n * DIM + k] = W16[i];
    }
    __syncthreads();

    const int wave = tid >> 5;
    const int lane = tid & 31;
    const int half = lane >> 4;   // K-half selector (ISA 7.12.2 A/B layouts)
    const int l16  = lane & 15;
    const long tileM = (long)blockIdx.x * 128 + (long)wave * 16;
    if (tileM >= M) return;

    long arow = tileM + l16;
    if (arow >= M) arow = M - 1;            // clamp reads; writes guarded below

    v8f acc[8];
#pragma unroll
    for (int n = 0; n < 8; ++n)
#pragma unroll
        for (int j = 0; j < 8; ++j) acc[n][j] = 0.0f;

#pragma unroll
    for (int kb = 0; kb < DIM; kb += 32) {
        // A fragment 16x32 f16: lanes 0-15 hold K kb+0..7 & kb+16..23,
        // lanes 16-31 hold K kb+8..15 & kb+24..31.
        const _Float16* ap = A16 + arow * DIM + kb + half * 8;
        v8h alo = *(const v8h*)(ap);
        v8h ahi = *(const v8h*)(ap + 16);
        v16h afrag;
#pragma unroll
        for (int i = 0; i < 8; ++i) { afrag[i] = alo[i]; afrag[i + 8] = ahi[i]; }

#pragma unroll
        for (int n = 0; n < 8; ++n) {
            // B fragment 32x16 f16: column = n*16 + l16; lanes 0-15 hold
            // K kb+0..7 & kb+16..23, lanes 16-31 hold K kb+8..15 & kb+24..31.
            const _Float16* bp = &Wt[(n * 16 + l16) * DIM + kb + half * 8];
            v8h b0 = *(const v8h*)(bp);
            v8h b1 = *(const v8h*)(bp + 16);
            v16h bfrag;
#pragma unroll
            for (int i = 0; i < 8; ++i) { bfrag[i] = b0[i]; bfrag[i + 8] = b1[i]; }

            acc[n] = __builtin_amdgcn_wmma_f32_16x16x32_f16(
                         /*neg_a=*/false, afrag, /*neg_b=*/false, bfrag,
                         /*c_mod=*/(short)0, acc[n],
                         /*reuse_a=*/false, /*reuse_b=*/false);
        }
    }

    // C/D layout: lane l -> col (l&15); rows m = (l>>4)*8 + v.
#pragma unroll
    for (int n = 0; n < 8; ++n) {
#pragma unroll
        for (int v = 0; v < 8; ++v) {
            long r = tileM + half * 8 + v;
            if (r < M) Out[r * DIM + n * 16 + l16] = acc[n][v];
        }
    }
}

// ---------------------------------------------------------------------------
// base[e,:] = (xu[row[e],:] + xv[col[e],:] + edge_attr[e,:] @ W_edge) / 3
// ---------------------------------------------------------------------------
__global__ __launch_bounds__(128)
void edge_base_kernel(const float* __restrict__ xu, const float* __restrict__ xv,
                      const float* __restrict__ ea, const float* __restrict__ We,
                      const int* __restrict__ row, const int* __restrict__ col,
                      float* __restrict__ base, long E)
{
    long e = blockIdx.x;
    if (e >= E) return;
    __shared__ float eatt[EDIM];
    int d = threadIdx.x;
    if (d < EDIM) eatt[d] = ea[e * EDIM + d];
    __syncthreads();
    float s = 0.0f;
#pragma unroll
    for (int k = 0; k < EDIM; ++k) s += eatt[k] * We[k * DIM + d];
    long r = row[e], c = col[e];
    base[e * DIM + d] = (xu[r * DIM + d] + xv[c * DIM + d] + s) * (1.0f / 3.0f);
}

// Line-graph scatter add: dst[bcol[j],:] += src[brow[j],:]
__global__ void scatter_add_edges(const float* __restrict__ src,
                                  const int* __restrict__ brow,
                                  const int* __restrict__ bcol,
                                  float* __restrict__ dst, long EB)
{
    long i = (long)blockIdx.x * blockDim.x + threadIdx.x;
    long total = EB * DIM;
    if (i >= total) return;
    long j = i >> 7;
    int  d = (int)(i & (DIM - 1));
    atomicAdd(&dst[(long)bcol[j] * DIM + d], src[(long)brow[j] * DIM + d]);
}

__global__ void add_store(const float* __restrict__ base, const float* __restrict__ agg,
                          float* __restrict__ out, long n)
{
    long i = (long)blockIdx.x * blockDim.x + threadIdx.x;
    if (i < n) out[i] = base[i] + agg[i];
}

// x_conv[e] = conv_agg[e]·w_rel + b_rel + out[e]·w_root ; atomicMax per graph.
// One wave32 per edge.
__global__ __launch_bounds__(256)
void xconv_kernel(const float* __restrict__ agg2, const float* __restrict__ out,
                  const float* __restrict__ w_rel, const float* __restrict__ b_rel,
                  const float* __restrict__ w_root, const int* __restrict__ batch,
                  float* __restrict__ xconv, unsigned* __restrict__ mkey, long E)
{
    long e = (long)blockIdx.x * 8 + (threadIdx.x >> 5);
    if (e >= E) return;
    int lane = threadIdx.x & 31;
    float s = 0.0f;
    for (int d = lane; d < DIM; d += 32)
        s += agg2[e * DIM + d] * w_rel[d] + out[e * DIM + d] * w_root[d];
#pragma unroll
    for (int off = 16; off > 0; off >>= 1) s += __shfl_down(s, off, 32);
    if (lane == 0) {
        float v = s + b_rel[0];
        xconv[e] = v;
        atomicMax(&mkey[batch[e]], fkey(v));
    }
}

__global__ void expdenom_kernel(const float* __restrict__ xconv,
                                const unsigned* __restrict__ mkey,
                                const int* __restrict__ batch,
                                float* __restrict__ eexp, float* __restrict__ denom, long E)
{
    long e = (long)blockIdx.x * blockDim.x + threadIdx.x;
    if (e >= E) return;
    float m = fval(mkey[batch[e]]);
    float v = expf(xconv[e] - m);
    eexp[e] = v;
    atomicAdd(&denom[batch[e]], v);
}

__global__ void scores_kernel(const float* __restrict__ eexp, const float* __restrict__ denom,
                              const int* __restrict__ batch, float* __restrict__ scores, long E)
{
    long e = (long)blockIdx.x * blockDim.x + threadIdx.x;
    if (e >= E) return;
    scores[e] = eexp[e] / (denom[batch[e]] + 1e-16f);
}

// gx[b,:] += out[e,:] * scores[e]
__global__ void gx_kernel(const float* __restrict__ out, const float* __restrict__ scores,
                          const int* __restrict__ batch, float* __restrict__ gx, long E)
{
    long i = (long)blockIdx.x * blockDim.x + threadIdx.x;
    long total = E * DIM;
    if (i >= total) return;
    long e = i >> 7;
    int  d = (int)(i & (DIM - 1));
    atomicAdd(&gx[(long)batch[e] * DIM + d], out[i] * scores[e]);
}

// gout[b,d] = tanh(gx[b,:] @ W_gout[:,d] + b_gout[d])
__global__ __launch_bounds__(128)
void gout_kernel(const float* __restrict__ gx, const float* __restrict__ Wg,
                 const float* __restrict__ bg, float* __restrict__ gout, int B)
{
    int b = blockIdx.x;
    if (b >= B) return;
    int d = threadIdx.x;
    __shared__ float gs[DIM];
    gs[d] = gx[(long)b * DIM + d];
    __syncthreads();
    float s = bg[d];
#pragma unroll 8
    for (int k = 0; k < DIM; ++k) s += gs[k] * Wg[k * DIM + d];
    gout[(long)b * DIM + d] = tanhf(s);
}

// Per-graph attention over the 3 iterations (softmax over T).
__global__ void sc_kernel(const float* __restrict__ gout_all,  // [T][B*DIM]
                          const float* __restrict__ a,         // [DIM*T], a[d*T+t]
                          const float* __restrict__ a_bias,    // [T]
                          float* __restrict__ sc, int B)
{
    int b = blockIdx.x * blockDim.x + threadIdx.x;
    if (b >= B) return;
    float s[T_ITER];
#pragma unroll
    for (int t = 0; t < T_ITER; ++t) {
        float acc = a_bias[t];
        const float* g = gout_all + (long)t * B * DIM + (long)b * DIM;
        for (int d = 0; d < DIM; ++d) acc += g[d] * a[d * T_ITER + t];
        s[t] = acc;
    }
    float m = fmaxf(s[0], fmaxf(s[1], s[2]));
    float e0 = expf(s[0] - m), e1 = expf(s[1] - m), e2 = expf(s[2] - m);
    float inv = 1.0f / (e0 + e1 + e2);
    sc[b * T_ITER + 0] = e0 * inv;
    sc[b * T_ITER + 1] = e1 * inv;
    sc[b * T_ITER + 2] = e2 * inv;
}

// y[col[e],:] += sum_t out_all[t][e,:] * sc[batch[e],t]
__global__ void final_kernel(const float* __restrict__ out_all,  // [T][E*DIM]
                             const float* __restrict__ sc,
                             const int* __restrict__ batch, const int* __restrict__ col,
                             float* __restrict__ y, long E)
{
    long i = (long)blockIdx.x * blockDim.x + threadIdx.x;
    long total = E * DIM;
    if (i >= total) return;
    long e = i >> 7;
    int  d = (int)(i & (DIM - 1));
    int  b = batch[e];
    float v = out_all[i]             * sc[b * T_ITER + 0]
            + out_all[total + i]     * sc[b * T_ITER + 1]
            + out_all[2 * total + i] * sc[b * T_ITER + 2];
    atomicAdd(&y[(long)col[e] * DIM + d], v);
}

// ---------------------------------------------------------------------------
static inline long cdiv(long a, long b) { return (a + b - 1) / b; }

extern "C" void kernel_launch(void* const* d_in, const int* in_sizes, int n_in,
                              void* d_out, int out_size, void* d_ws, size_t ws_size,
                              hipStream_t stream)
{
    const float* x         = (const float*)d_in[0];
    const float* edge_attr = (const float*)d_in[1];
    const float* W_u       = (const float*)d_in[2];
    const float* W_v       = (const float*)d_in[3];
    const float* W_edge    = (const float*)d_in[4];
    const float* w_rel     = (const float*)d_in[5];
    const float* b_rel     = (const float*)d_in[6];
    const float* w_root    = (const float*)d_in[7];
    const float* W_gout    = (const float*)d_in[8];
    const float* b_gout    = (const float*)d_in[9];
    const float* a_att     = (const float*)d_in[10];
    const float* a_bias    = (const float*)d_in[11];
    const int*   edge_idx  = (const int*)d_in[12];
    const int*   bond_idx  = (const int*)d_in[13];
    const int*   batch     = (const int*)d_in[14];

    const long N  = in_sizes[0] / DIM;
    const long E  = in_sizes[1] / EDIM;
    const long EB = in_sizes[13] / 2;
    const int  B  = NGRAPH;

    const int* row  = edge_idx;
    const int* col  = edge_idx + E;
    const int* brow = bond_idx;
    const int* bcol = bond_idx + EB;

    // ---- workspace carve (256B aligned) ----
    char*  wsp = (char*)d_ws;
    size_t off = 0;
    auto carve = [&](size_t bytes) -> void* {
        void* p = wsp + off;
        off += (bytes + 255) & ~(size_t)255;
        return p;
    };
    _Float16* x16   = (_Float16*)carve((size_t)N * DIM * sizeof(_Float16));
    _Float16* Wu16  = (_Float16*)carve((size_t)DIM * DIM * sizeof(_Float16));
    _Float16* Wv16  = (_Float16*)carve((size_t)DIM * DIM * sizeof(_Float16));
    float* xu       = (float*)carve((size_t)N * DIM * sizeof(float));
    float* xv       = (float*)carve((size_t)N * DIM * sizeof(float));
    float* base     = (float*)carve((size_t)E * DIM * sizeof(float));
    float* out_all  = (float*)carve((size_t)T_ITER * E * DIM * sizeof(float));
    float* agg      = (float*)carve((size_t)E * DIM * sizeof(float));
    float* xconv    = (float*)carve((size_t)E * sizeof(float));
    float* eexp     = (float*)carve((size_t)E * sizeof(float));
    float* scores   = (float*)carve((size_t)E * sizeof(float));
    unsigned* mkey  = (unsigned*)carve((size_t)B * sizeof(unsigned));
    float* denom    = (float*)carve((size_t)B * sizeof(float));
    float* gx       = (float*)carve((size_t)B * DIM * sizeof(float));
    float* gout_all = (float*)carve((size_t)T_ITER * B * DIM * sizeof(float));
    float* sc       = (float*)carve((size_t)B * T_ITER * sizeof(float));
    (void)ws_size; (void)n_in; (void)out_size;

    const int TB = 256;

    // ---- precision staging: f32 -> f16 for WMMA operands ----
    f32_to_f16<<<cdiv(N * DIM, TB), TB, 0, stream>>>(x,   x16,  N * DIM);
    f32_to_f16<<<cdiv(DIM * DIM, TB), TB, 0, stream>>>(W_u, Wu16, DIM * DIM);
    f32_to_f16<<<cdiv(DIM * DIM, TB), TB, 0, stream>>>(W_v, Wv16, DIM * DIM);

    // ---- node GEMMs on WMMA ----
    long gemm_blocks = cdiv(N, 128);
    node_gemm_wmma<<<gemm_blocks, 256, 0, stream>>>(x16, Wu16, xu, (int)N);
    node_gemm_wmma<<<gemm_blocks, 256, 0, stream>>>(x16, Wv16, xv, (int)N);

    // ---- per-edge base message ----
    edge_base_kernel<<<E, DIM, 0, stream>>>(xu, xv, edge_attr, W_edge, row, col, base, E);

    // ---- 3 message-passing iterations over the line graph ----
    const float* src = base;   // out before iteration 0 == base
    for (int t = 0; t < T_ITER; ++t) {
        float* cur = out_all + (long)t * E * DIM;

        zero_f32<<<cdiv(E * DIM, TB), TB, 0, stream>>>(agg, E * DIM);
        scatter_add_edges<<<cdiv(EB * DIM, TB), TB, 0, stream>>>(src, brow, bcol, agg, EB);
        add_store<<<cdiv(E * DIM, TB), TB, 0, stream>>>(base, agg, cur, E * DIM);

        zero_f32<<<cdiv(E * DIM, TB), TB, 0, stream>>>(agg, E * DIM);
        scatter_add_edges<<<cdiv(EB * DIM, TB), TB, 0, stream>>>(cur, brow, bcol, agg, EB);

        set_u32<<<cdiv(B, TB), TB, 0, stream>>>(mkey, FKEY_NEG_INF, B);
        zero_f32<<<cdiv(B, TB), TB, 0, stream>>>(denom, B);
        zero_f32<<<cdiv((long)B * DIM, TB), TB, 0, stream>>>(gx, (long)B * DIM);

        xconv_kernel<<<cdiv(E, 8), 256, 0, stream>>>(agg, cur, w_rel, b_rel, w_root,
                                                     batch, xconv, mkey, E);
        expdenom_kernel<<<cdiv(E, TB), TB, 0, stream>>>(xconv, mkey, batch, eexp, denom, E);
        scores_kernel<<<cdiv(E, TB), TB, 0, stream>>>(eexp, denom, batch, scores, E);
        gx_kernel<<<cdiv(E * DIM, TB), TB, 0, stream>>>(cur, scores, batch, gx, E);
        gout_kernel<<<B, DIM, 0, stream>>>(gx, W_gout, b_gout,
                                           gout_all + (long)t * B * DIM, B);
        src = cur;
    }

    // ---- iteration attention + final node scatter ----
    sc_kernel<<<cdiv(B, TB), TB, 0, stream>>>(gout_all, a_att, a_bias, sc, B);
    hipMemcpyAsync(d_out, x, (size_t)N * DIM * sizeof(float),
                   hipMemcpyDeviceToDevice, stream);
    final_kernel<<<cdiv(E * DIM, TB), TB, 0, stream>>>(out_all, sc, batch, col,
                                                       (float*)d_out, E);
}
